// RNNSearch_19945828122798
// MI455X (gfx1250) — matrix operand order repaired and checked by reading, not compiled, not run on hardware
//
#include <hip/hip_runtime.h>
#include <math.h>

#define N_VOCAB 32000
#define D_EMBED 512
#define D_ENC   512
#define D_DEC   1024
#define BSZ     32
#define SRC_LEN 128
#define STEPS   48

typedef __bf16 bf16;
typedef __attribute__((ext_vector_type(16))) bf16  v16bf;
typedef __attribute__((ext_vector_type(8)))  float v8f;

// Async copy 16 bytes global -> LDS (per lane), tracked by ASYNCcnt.
__device__ __forceinline__ void async_ld_b128(unsigned lds_off, const void* gptr) {
  asm volatile("global_load_async_to_lds_b128 %0, %1, off"
               :: "v"(lds_off), "v"((unsigned long long)(uintptr_t)gptr)
               : "memory");
}
__device__ __forceinline__ void wait_async0() {
  asm volatile("s_wait_asynccnt 0" ::: "memory");
}

// ---------------------------------------------------------------------------
// Generic bf16 GEMM: C(MxN) = A(MxK) @ W(NxK)^T + bias(N), C fp32.
// Requires M%32==0, N%128==0, K%32==0 (true for every matmul in this model).
// Workgroup = 128 threads = 4 waves, one per 32-wide N strip; each wave owns
// a 32x32 output tile: 2 A fragments x 2 B fragments -> 4 WMMA accumulators,
// so every fragment fetched from LDS feeds two v_wmma issues.
// All staging is async global->LDS (b128), double-buffered: tile k+1 streams
// in while tile k is computed; one s_wait_asynccnt + barrier per K-step.
// ---------------------------------------------------------------------------
__global__ __launch_bounds__(128) void k_gemm_bf16(
    const bf16* __restrict__ A, int lda,
    const bf16* __restrict__ W, int ldw,
    const float* __restrict__ bias,
    float* __restrict__ C, long ldc,
    int M, int N, int K)
{
  __shared__ __align__(16) bf16 sA[2][32][40];
  __shared__ __align__(16) bf16 sB[2][128][40];

  const int tid  = threadIdx.x;
  const int n0   = blockIdx.x * 128;
  const int m0   = blockIdx.y * 32;
  const int wn   = tid >> 5;    // wave id = N strip (0..3)
  const int lane = tid & 31;
  const int fm   = lane & 15;
  const int fh   = lane >> 4;

  v8f acc00, acc01, acc10, acc11;
  const float bv0 = bias ? bias[n0 + wn * 32 + fm]      : 0.0f;
  const float bv1 = bias ? bias[n0 + wn * 32 + 16 + fm] : 0.0f;
#pragma unroll
  for (int j = 0; j < 8; ++j) {
    acc00[j] = bv0; acc01[j] = bv1; acc10[j] = bv0; acc11[j] = bv1;
  }

  const int sr = tid >> 2;        // 0..31 (staging row)
  const int sc = (tid & 3) * 8;   // 0..24 (8 bf16 = 16 bytes per thread)

  // Stage one 32-wide K slice into LDS buffer `buf` (fully async).
  auto stage = [&](int k0, int buf) {
    async_ld_b128((unsigned)(uintptr_t)&sA[buf][sr][sc],
                  A + (size_t)(m0 + sr) * lda + k0 + sc);
#pragma unroll
    for (int i = 0; i < 4; ++i) {
      const int br = sr + i * 32;
      async_ld_b128((unsigned)(uintptr_t)&sB[buf][br][sc],
                    W + (size_t)(n0 + br) * ldw + k0 + sc);
    }
    // keep L2 one more tile ahead on the streaming weight operand
    if (k0 + 32 < K)
      __builtin_prefetch(W + (size_t)(n0 + sr) * ldw + k0 + 32 + sc, 0, 1);
  };

  stage(0, 0);
  wait_async0();
  __syncthreads();

  int buf = 0;
  for (int k0 = 0; k0 < K; k0 += 32) {
    const bool more = (k0 + 32) < K;
    if (more) stage(k0 + 32, buf ^ 1);

    // A fragments (16x32 bf16 each, ISA 16-bit A layout): rows 0-15 / 16-31.
    v16bf a0, a1, b0, b1;
#pragma unroll
    for (int i = 0; i < 8; ++i) {
      a0[i]     = sA[buf][fm][fh * 8 + i];
      a0[8 + i] = sA[buf][fm][16 + fh * 8 + i];
      a1[i]     = sA[buf][16 + fm][fh * 8 + i];
      a1[8 + i] = sA[buf][16 + fm][16 + fh * 8 + i];
    }
    // B fragments (32x16 each): same K order as A (lanes 0-15 K=0..7,16..23; lanes 16-31 K=8..15,24..31).
#pragma unroll
    for (int i = 0; i < 8; ++i) {
      b0[i]     = sB[buf][wn * 32 + fm][fh * 8 + i];
      b0[8 + i] = sB[buf][wn * 32 + fm][16 + fh * 8 + i];
      b1[i]     = sB[buf][wn * 32 + 16 + fm][fh * 8 + i];
      b1[8 + i] = sB[buf][wn * 32 + 16 + fm][16 + fh * 8 + i];
    }

    acc00 = __builtin_amdgcn_wmma_f32_16x16x32_bf16(
        false, a0, false, b0, (short)0, acc00, false, false);
    acc01 = __builtin_amdgcn_wmma_f32_16x16x32_bf16(
        false, a0, false, b1, (short)0, acc01, false, false);
    acc10 = __builtin_amdgcn_wmma_f32_16x16x32_bf16(
        false, a1, false, b0, (short)0, acc10, false, false);
    acc11 = __builtin_amdgcn_wmma_f32_16x16x32_bf16(
        false, a1, false, b1, (short)0, acc11, false, false);

    if (more) {
      wait_async0();      // next tile's async writes landed in LDS
      __syncthreads();    // publish across waves; fragment reads already done
    }
    buf ^= 1;
  }

#pragma unroll
  for (int j = 0; j < 8; ++j) {
    const int gm0 = m0 + fh * 8 + j;        // rows 0..15
    const int gm1 = m0 + 16 + fh * 8 + j;   // rows 16..31
    const int gn0 = n0 + wn * 32 + fm;
    C[(size_t)gm0 * ldc + gn0]      = acc00[j];
    C[(size_t)gm0 * ldc + gn0 + 16] = acc01[j];
    C[(size_t)gm1 * ldc + gn0]      = acc10[j];
    C[(size_t)gm1 * ldc + gn0 + 16] = acc11[j];
  }
}

// ---------------------------------------------------------------------------
// Elementwise / gather / reduction helpers
// ---------------------------------------------------------------------------
__global__ void k_cvt(const float* __restrict__ s, bf16* __restrict__ d, long n) {
  long i = (long)blockIdx.x * blockDim.x + threadIdx.x;
  if (i < n) d[i] = (bf16)s[i];
}

__global__ void k_zero2(float* __restrict__ h, bf16* __restrict__ hb, int n) {
  int i = blockIdx.x * blockDim.x + threadIdx.x;
  if (i < n) { h[i] = 0.0f; hb[i] = (bf16)0.0f; }
}

// x[(t*B+b)*E + e] = tok_emb_bf[ids[b*T+t]*E + e]
__global__ void k_embed_seq(const int* __restrict__ ids,
                            const bf16* __restrict__ emb,
                            bf16* __restrict__ x) {
  long i = (long)blockIdx.x * blockDim.x + threadIdx.x;
  if (i >= (long)SRC_LEN * BSZ * D_EMBED) return;
  int e = (int)(i % D_EMBED);
  long tb = i / D_EMBED;
  int t = (int)(tb / BSZ);
  int b = (int)(tb % BSZ);
  x[i] = emb[(size_t)ids[b * SRC_LEN + t] * D_EMBED + e];
}

// GRU gate fusion: r,z,n gates -> new h (fp32 + bf16 copy), optional y write.
__global__ void k_gru_fuse(const float* __restrict__ gi,
                           const float* __restrict__ gh,
                           float* __restrict__ h, bf16* __restrict__ hb,
                           int H, bf16* __restrict__ y, int ldy) {
  int i = blockIdx.x * blockDim.x + threadIdx.x;
  if (i >= BSZ * H) return;
  int b = i / H, j = i % H;
  const float* gib = gi + (size_t)b * 3 * H;
  const float* ghb = gh + (size_t)b * 3 * H;
  float r  = 1.0f / (1.0f + expf(-(gib[j]         + ghb[j])));
  float zt = 1.0f / (1.0f + expf(-(gib[H + j]     + ghb[H + j])));
  float nn = tanhf(gib[2 * H + j] + r * ghb[2 * H + j]);
  float hv = (1.0f - zt) * nn + zt * h[i];
  h[i] = hv;
  hb[i] = (bf16)hv;
  if (y) y[(size_t)b * ldy + j] = (bf16)hv;
}

// h0_cat assemble with ReLU: order (l0f, l0b, l1f, l1b), each 512 wide.
__global__ void k_h0cat(const float* __restrict__ h00, const float* __restrict__ h01,
                        const float* __restrict__ h10, const float* __restrict__ h11,
                        bf16* __restrict__ out) {
  int i = blockIdx.x * blockDim.x + threadIdx.x;
  if (i >= BSZ * 2048) return;
  int b = i / 2048, j = i % 2048;
  int which = j >> 9, jj = j & 511;
  const float* src = which == 0 ? h00 : which == 1 ? h01 : which == 2 ? h10 : h11;
  float v = src[b * 512 + jj];
  out[i] = (bf16)fmaxf(v, 0.0f);
}

// Decoder initial state from style embedding.
__global__ void k_style(const int* __restrict__ labels, const float* __restrict__ se,
                        float* __restrict__ h0, float* __restrict__ h1,
                        bf16* __restrict__ h0b, bf16* __restrict__ h1b) {
  int i = blockIdx.x * blockDim.x + threadIdx.x;
  if (i >= BSZ * 2 * D_DEC) return;
  int b = i / (2 * D_DEC), j = i % (2 * D_DEC);
  float v = se[(size_t)labels[b] * 2 * D_DEC + j];
  int l = j >> 10, jj = j & 1023;
  if (l == 0) { h0[b * D_DEC + jj] = v; h0b[b * D_DEC + jj] = (bf16)v; }
  else        { h1[b * D_DEC + jj] = v; h1b[b * D_DEC + jj] = (bf16)v; }
}

__global__ void k_start(const float* __restrict__ st, bf16* __restrict__ eb) {
  int i = blockIdx.x * blockDim.x + threadIdx.x;
  if (i >= BSZ * D_EMBED) return;
  eb[i] = (bf16)st[i % D_EMBED];
}

// scores[b][t] = scale * dot(q[b], memory[t][b])  (memory time-major, bf16)
__global__ __launch_bounds__(128) void k_attn_scores(
    const float* __restrict__ q, const bf16* __restrict__ mem,
    float* __restrict__ scores, float scale) {
  int bt = blockIdx.x;
  int b = bt / SRC_LEN, t = bt % SRC_LEN;
  __shared__ float red[128];
  const float* qb = q + (size_t)b * 1024;
  const bf16* m = mem + ((size_t)t * BSZ + b) * 1024;
  float s = 0.0f;
  for (int d = threadIdx.x; d < 1024; d += 128) s += qb[d] * (float)m[d];
  red[threadIdx.x] = s;
  __syncthreads();
  for (int off = 64; off > 0; off >>= 1) {
    if (threadIdx.x < off) red[threadIdx.x] += red[threadIdx.x + off];
    __syncthreads();
  }
  if (threadIdx.x == 0) scores[b * SRC_LEN + t] = red[0] * scale;
}

__global__ __launch_bounds__(128) void k_softmax(float* __restrict__ scores) {
  int b = blockIdx.x, t = threadIdx.x;
  __shared__ float red[128];
  float v = scores[b * SRC_LEN + t];
  red[t] = v;
  __syncthreads();
  for (int off = 64; off > 0; off >>= 1) {
    if (t < off) red[t] = fmaxf(red[t], red[t + off]);
    __syncthreads();
  }
  float mx = red[0];
  __syncthreads();
  float e = expf(v - mx);
  red[t] = e;
  __syncthreads();
  for (int off = 64; off > 0; off >>= 1) {
    if (t < off) red[t] += red[t + off];
    __syncthreads();
  }
  scores[b * SRC_LEN + t] = e / red[0];
}

__global__ __launch_bounds__(256) void k_attn_ctx(
    const float* __restrict__ a, const bf16* __restrict__ mem,
    float* __restrict__ c) {
  int b = blockIdx.x;
  for (int d = threadIdx.x; d < 1024; d += 256) {
    float s = 0.0f;
    for (int t = 0; t < SRC_LEN; ++t)
      s += a[b * SRC_LEN + t] * (float)mem[((size_t)t * BSZ + b) * 1024 + d];
    c[(size_t)b * 1024 + d] = s;
  }
}

// xin[b][0:512] = embed_bf[b], xin[b][512:1536] = (bf16)ctx[b]
__global__ void k_concat(const bf16* __restrict__ eb, const float* __restrict__ c,
                         bf16* __restrict__ xin) {
  int i = blockIdx.x * blockDim.x + threadIdx.x;
  if (i >= BSZ * 1536) return;
  int b = i / 1536, j = i % 1536;
  xin[i] = (j < 512) ? eb[b * 512 + j] : (bf16)c[(size_t)b * 1024 + (j - 512)];
}

__global__ __launch_bounds__(256) void k_argmax(
    const float* __restrict__ logits, long ld, int* __restrict__ idx) {
  int b = blockIdx.x;
  const float* row = logits + (size_t)b * ld;
  float best = -INFINITY; int bi = 0;
  for (int v = threadIdx.x; v < N_VOCAB; v += 256) {
    float x = row[v];
    if (x > best) { best = x; bi = v; }
  }
  __shared__ float bv[256];
  __shared__ int bx[256];
  bv[threadIdx.x] = best; bx[threadIdx.x] = bi;
  __syncthreads();
  for (int off = 128; off > 0; off >>= 1) {
    if (threadIdx.x < off) {
      float o = bv[threadIdx.x + off]; int oi = bx[threadIdx.x + off];
      if (o > bv[threadIdx.x] || (o == bv[threadIdx.x] && oi < bx[threadIdx.x])) {
        bv[threadIdx.x] = o; bx[threadIdx.x] = oi;
      }
    }
    __syncthreads();
  }
  if (threadIdx.x == 0) idx[b] = bx[0];
}

__global__ void k_embed_step(const int* __restrict__ idx,
                             const bf16* __restrict__ emb,
                             bf16* __restrict__ eb) {
  int i = blockIdx.x * blockDim.x + threadIdx.x;
  if (i >= BSZ * D_EMBED) return;
  int b = i / D_EMBED, e = i % D_EMBED;
  eb[i] = emb[(size_t)idx[b] * D_EMBED + e];
}

// ---------------------------------------------------------------------------
// Host orchestration
// ---------------------------------------------------------------------------
extern "C" void kernel_launch(void* const* d_in, const int* in_sizes, int n_in,
                              void* d_out, int out_size, void* d_ws, size_t ws_size,
                              hipStream_t stream) {
  (void)in_sizes; (void)n_in; (void)out_size; (void)ws_size;

  const int*   input_ids = (const int*)d_in[0];
  const int*   labels    = (const int*)d_in[1];
  // d_in[2] = max_steps (device scalar; reference value 48, fixed for determinism)
  const float* tok_emb   = (const float*)d_in[3];
  const float* style_emb = (const float*)d_in[4];
  const float* start_emb = (const float*)d_in[5];
  int pi = 6;
  const float *eWih[2][2], *eWhh[2][2], *eBih[2][2], *eBhh[2][2];
  for (int l = 0; l < 2; ++l)
    for (int d = 0; d < 2; ++d) {
      eWih[l][d] = (const float*)d_in[pi++];
      eWhh[l][d] = (const float*)d_in[pi++];
      eBih[l][d] = (const float*)d_in[pi++];
      eBhh[l][d] = (const float*)d_in[pi++];
    }
  const float *dWih[2], *dWhh[2], *dBih[2], *dBhh[2];
  for (int l = 0; l < 2; ++l) {
    dWih[l] = (const float*)d_in[pi++];
    dWhh[l] = (const float*)d_in[pi++];
    dBih[l] = (const float*)d_in[pi++];
    dBhh[l] = (const float*)d_in[pi++];
  }
  const float* Wt  = (const float*)d_in[pi++];
  const float* bt  = (const float*)d_in[pi++];
  const float* Wq  = (const float*)d_in[pi++];
  const float* bq  = (const float*)d_in[pi++];
  const float* Wf  = (const float*)d_in[pi++];
  const float* bff = (const float*)d_in[pi++];
  const float* Wv  = (const float*)d_in[pi++];
  const float* bvv = (const float*)d_in[pi++];

  // workspace allocation
  char* base = (char*)d_ws;
  size_t off = 0;
  auto alloc = [&](size_t bytes) -> void* {
    void* p = base + off;
    off = (off + bytes + 255) & ~(size_t)255;
    return p;
  };
  auto allocB = [&](size_t elems) -> bf16*  { return (bf16*)alloc(elems * 2); };
  auto allocF = [&](size_t elems) -> float* { return (float*)alloc(elems * 4); };

  bf16* tokE_b = allocB((size_t)N_VOCAB * D_EMBED);
  bf16 *eWih_b[2][2], *eWhh_b[2][2];
  for (int l = 0; l < 2; ++l)
    for (int d = 0; d < 2; ++d) {
      int din = (l == 0) ? D_EMBED : 2 * D_ENC;
      eWih_b[l][d] = allocB((size_t)3 * D_ENC * din);
      eWhh_b[l][d] = allocB((size_t)3 * D_ENC * D_ENC);
    }
  bf16 *dWih_b[2], *dWhh_b[2];
  for (int l = 0; l < 2; ++l) {
    dWih_b[l] = allocB((size_t)3 * D_DEC * D_DEC);
    dWhh_b[l] = allocB((size_t)3 * D_DEC * D_DEC);
  }
  bf16* Wt_b = allocB((size_t)D_DEC * 2048);
  bf16* Wq_b = allocB((size_t)1024 * D_DEC);
  bf16* Wf_b = allocB((size_t)D_DEC * 1536);
  bf16* Wv_b = allocB((size_t)N_VOCAB * D_DEC);

  bf16* x0 = allocB((size_t)SRC_LEN * BSZ * D_EMBED);
  bf16* x1 = allocB((size_t)SRC_LEN * BSZ * 1024);
  bf16* x2 = allocB((size_t)SRC_LEN * BSZ * 1024);   // memory
  float* giBig = allocF((size_t)SRC_LEN * BSZ * 1536);
  float* ghBuf = allocF((size_t)BSZ * 3072);
  float* giDec = allocF((size_t)BSZ * 3072);
  float *hE[2][2]; bf16* hEb[2][2];
  for (int l = 0; l < 2; ++l)
    for (int d = 0; d < 2; ++d) {
      hE[l][d]  = allocF((size_t)BSZ * D_ENC);
      hEb[l][d] = allocB((size_t)BSZ * D_ENC);
    }
  bf16* h0cat_b = allocB((size_t)BSZ * 2048);
  float* out0   = allocF((size_t)BSZ * D_DEC);
  bf16*  out0_b = allocB((size_t)BSZ * D_DEC);
  float* qv     = allocF((size_t)BSZ * 1024);
  float* scores = allocF((size_t)BSZ * SRC_LEN);
  float* ctx    = allocF((size_t)BSZ * 1024);
  bf16*  xin_b  = allocB((size_t)BSZ * 1536);
  float* xd     = allocF((size_t)BSZ * D_DEC);
  bf16*  xd_b   = allocB((size_t)BSZ * D_DEC);
  float* hD[2]; bf16* hDb[2];
  for (int l = 0; l < 2; ++l) {
    hD[l]  = allocF((size_t)BSZ * D_DEC);
    hDb[l] = allocB((size_t)BSZ * D_DEC);
  }
  int*  idxB  = (int*)alloc(BSZ * 4);
  bf16* emb_b = allocB((size_t)BSZ * D_EMBED);

  auto cvt = [&](const float* s, bf16* d, size_t n) {
    k_cvt<<<dim3((unsigned)((n + 255) / 256)), dim3(256), 0, stream>>>(s, d, (long)n);
  };
  auto gemm = [&](const bf16* A, int lda, const bf16* W, int ldw,
                  const float* bias, float* C, long ldc, int M, int N, int K) {
    k_gemm_bf16<<<dim3(N / 128, M / 32), dim3(128), 0, stream>>>(
        A, lda, W, ldw, bias, C, ldc, M, N, K);
  };

  // ---- one-time weight conversions (idempotent per call) ----
  cvt(tok_emb, tokE_b, (size_t)N_VOCAB * D_EMBED);
  for (int l = 0; l < 2; ++l)
    for (int d = 0; d < 2; ++d) {
      int din = (l == 0) ? D_EMBED : 2 * D_ENC;
      cvt(eWih[l][d], eWih_b[l][d], (size_t)3 * D_ENC * din);
      cvt(eWhh[l][d], eWhh_b[l][d], (size_t)3 * D_ENC * D_ENC);
    }
  for (int l = 0; l < 2; ++l) {
    cvt(dWih[l], dWih_b[l], (size_t)3 * D_DEC * D_DEC);
    cvt(dWhh[l], dWhh_b[l], (size_t)3 * D_DEC * D_DEC);
  }
  cvt(Wt, Wt_b, (size_t)D_DEC * 2048);
  cvt(Wq, Wq_b, (size_t)1024 * D_DEC);
  cvt(Wf, Wf_b, (size_t)D_DEC * 1536);
  cvt(Wv, Wv_b, (size_t)N_VOCAB * D_DEC);

  // ---- encoder ----
  {
    long n = (long)SRC_LEN * BSZ * D_EMBED;
    k_embed_seq<<<dim3((unsigned)((n + 255) / 256)), dim3(256), 0, stream>>>(
        input_ids, tokE_b, x0);
  }
  const bf16* xin = x0;
  int din = D_EMBED;
  for (int l = 0; l < 2; ++l) {
    bf16* xout = (l == 0) ? x1 : x2;
    for (int d = 0; d < 2; ++d) {
      // all-timestep input gates: (T*B, 3H) = x @ Wih^T + bih
      gemm(xin, din, eWih_b[l][d], din, eBih[l][d], giBig, 1536,
           SRC_LEN * BSZ, 1536, din);
      k_zero2<<<dim3(64), dim3(256), 0, stream>>>(hE[l][d], hEb[l][d], BSZ * D_ENC);
      for (int s = 0; s < SRC_LEN; ++s) {
        int t = (d == 0) ? s : (SRC_LEN - 1 - s);
        gemm(hEb[l][d], D_ENC, eWhh_b[l][d], D_ENC, eBhh[l][d], ghBuf, 1536,
             BSZ, 1536, D_ENC);
        k_gru_fuse<<<dim3((BSZ * D_ENC + 255) / 256), dim3(256), 0, stream>>>(
            giBig + (size_t)t * BSZ * 1536, ghBuf, hE[l][d], hEb[l][d], D_ENC,
            xout + (size_t)t * BSZ * 1024 + d * D_ENC, 1024);
      }
    }
    xin = xout;
    din = 1024;
  }

  // ---- decoder init ----
  k_h0cat<<<dim3((BSZ * 2048 + 255) / 256), dim3(256), 0, stream>>>(
      hE[0][0], hE[0][1], hE[1][0], hE[1][1], h0cat_b);
  gemm(h0cat_b, 2048, Wt_b, 2048, bt, out0, D_DEC, BSZ, D_DEC, 2048);
  cvt(out0, out0_b, (size_t)BSZ * D_DEC);
  k_style<<<dim3((BSZ * 2 * D_DEC + 255) / 256), dim3(256), 0, stream>>>(
      labels, style_emb, hD[0], hD[1], hDb[0], hDb[1]);
  k_start<<<dim3((BSZ * D_EMBED + 255) / 256), dim3(256), 0, stream>>>(
      start_emb, emb_b);

  const float scale = 0.03125f;  // 1/sqrt(2*D_ENC)
  const bf16* outq = out0_b;
  for (int step = 0; step < STEPS; ++step) {
    gemm(outq, D_DEC, Wq_b, D_DEC, bq, qv, 1024, BSZ, 1024, D_DEC);
    k_attn_scores<<<dim3(BSZ * SRC_LEN), dim3(128), 0, stream>>>(qv, x2, scores, scale);
    k_softmax<<<dim3(BSZ), dim3(128), 0, stream>>>(scores);
    k_attn_ctx<<<dim3(BSZ), dim3(256), 0, stream>>>(scores, x2, ctx);
    k_concat<<<dim3((BSZ * 1536 + 255) / 256), dim3(256), 0, stream>>>(
        emb_b, ctx, xin_b);
    gemm(xin_b, 1536, Wf_b, 1536, bff, xd, D_DEC, BSZ, D_DEC, 1536);
    cvt(xd, xd_b, (size_t)BSZ * D_DEC);

    const bf16* gin = xd_b;
    for (int l = 0; l < 2; ++l) {
      gemm(gin, D_DEC, dWih_b[l], D_DEC, dBih[l], giDec, 3072, BSZ, 3072, D_DEC);
      gemm(hDb[l], D_DEC, dWhh_b[l], D_DEC, dBhh[l], ghBuf, 3072, BSZ, 3072, D_DEC);
      k_gru_fuse<<<dim3((BSZ * D_DEC + 255) / 256), dim3(256), 0, stream>>>(
          giDec, ghBuf, hD[l], hDb[l], D_DEC, (bf16*)nullptr, 0);
      gin = hDb[l];
    }

    float* logits = (float*)d_out + (size_t)step * N_VOCAB;
    gemm(hDb[1], D_DEC, Wv_b, D_DEC, bvv, logits, (long)STEPS * N_VOCAB,
         BSZ, N_VOCAB, D_DEC);
    k_argmax<<<dim3(BSZ), dim3(256), 0, stream>>>(logits, (long)STEPS * N_VOCAB, idxB);
    k_embed_step<<<dim3((BSZ * D_EMBED + 255) / 256), dim3(256), 0, stream>>>(
        idxB, tokE_b, emb_b);
    outq = hDb[1];
  }
}
